// HPUMLAImpl_71141838291322
// MI455X (gfx1250) — matrix operand order repaired and checked by reading, not compiled, not run on hardware
//
#include <hip/hip_runtime.h>
#include <hip/hip_bf16.h>

typedef __attribute__((ext_vector_type(16))) _Float16 v16h;
typedef __attribute__((ext_vector_type(8)))  _Float16 v8h;
typedef __attribute__((ext_vector_type(8)))  float    v8f;

#define B_    2
#define S_    2048
#define H_    16
#define RANK_ 512
#define ROPE_ 64
#define NOPE_ 128
#define VD_   128
#define QK_   192          // NOPE + ROPE
#define LAT_  576          // RANK + ROPE
#define M_    4096         // B*S
#define N_    4096         // H*(NOPE+VD)
#define SCALE_ 0.07216878364870323f   // 1/sqrt(192)

__device__ __forceinline__ v16h pack8(v8h lo, v8h hi) {
  v16h r;
#pragma unroll
  for (int i = 0; i < 8; ++i) { r[i] = lo[i]; r[i + 8] = hi[i]; }
  return r;
}

__device__ __forceinline__ v8f wmma_f16(v16h a, v16h b, v8f c) {
  // D = A(16x32 f16) x B(32x16 f16) + C(16x16 f32)
  return __builtin_amdgcn_wmma_f32_16x16x32_f16(
      /*neg_a=*/false, a, /*neg_b=*/false, b,
      /*c_mod=*/(short)0, c, /*reuse_a=*/false, /*reuse_b=*/false);
}

// 16B async global -> LDS copy (ASYNCcnt-tracked, GV addressing)
__device__ __forceinline__ void async_ld16(unsigned lds_off, const void* gaddr) {
  asm volatile("global_load_async_to_lds_b128 %0, %1, off"
               :: "v"(lds_off), "v"(gaddr) : "memory");
}
__device__ __forceinline__ void wait_async0() {
  asm volatile("s_wait_asynccnt 0x0" ::: "memory");
}
__device__ __forceinline__ unsigned lds_off_of(const void* p) {
  return (unsigned)(unsigned long long)p;   // low 32 bits of LDS aperture addr
}

// ---------------- convert kernels ----------------

// latent[:, :512] (f32) -> Af16 [M_][RANK_] row-major f16
__global__ __launch_bounds__(256) void conv_a_kernel(
    const float* __restrict__ latent, _Float16* __restrict__ Af) {
  int i = blockIdx.x * 256 + threadIdx.x;        // M_*RANK_ exact
  int t = i >> 9, k = i & 511;
  Af[i] = (_Float16)latent[(size_t)t * LAT_ + k];
}

// W[k][n] (f32, 512x4096) -> Wt[n][k] f16 (coalesced read)
__global__ __launch_bounds__(256) void conv_w_kernel(
    const float* __restrict__ W, _Float16* __restrict__ Wt) {
  int i = blockIdx.x * 256 + threadIdx.x;        // RANK_*N_ exact
  int k = i >> 12, n = i & 4095;
  Wt[(size_t)n * RANK_ + k] = (_Float16)W[i];
}

// broadcast k_pe (latent[:,512:576]) into Kbuf[b][h][s][128+r] for all h
__global__ __launch_bounds__(256) void conv_pe_kernel(
    const float* __restrict__ latent, _Float16* __restrict__ Kbuf) {
  int i = blockIdx.x * 256 + threadIdx.x;        // M_*H_*ROPE_ exact
  int r = i & 63;
  int h = (i >> 6) & 15;
  int t = i >> 10;
  int b = t >> 11, s = t & 2047;
  _Float16 v = (_Float16)latent[(size_t)t * LAT_ + RANK_ + r];
  Kbuf[((size_t)(b * H_ + h) * S_ + s) * QK_ + NOPE_ + r] = v;
}

// ---------------- KV up-projection GEMM ----------------
__global__ __launch_bounds__(256) void kv_gemm_kernel(
    const _Float16* __restrict__ Af, const _Float16* __restrict__ Wt,
    _Float16* __restrict__ Kbuf, _Float16* __restrict__ Vt) {
  const int lane = threadIdx.x & 31;
  const int wid  = threadIdx.x >> 5;
  const int half = lane >> 4;
  const int l16  = lane & 15;
  const int wm = wid & 3;          // 4 waves along M
  const int wn = wid >> 2;         // 2 waves along N
  const int m0 = blockIdx.y * 128 + wm * 32;
  const int n0 = blockIdx.x * 128 + wn * 64;

  v8f acc[2][4] = {};

  for (int kc = 0; kc < RANK_; kc += 32) {
    v16h a[2];
#pragma unroll
    for (int fm = 0; fm < 2; ++fm) {
      const _Float16* p = Af + (size_t)(m0 + 16 * fm + l16) * RANK_ + kc + 8 * half;
      a[fm] = pack8(*(const v8h*)p, *(const v8h*)(p + 16));
    }
#pragma unroll
    for (int fn = 0; fn < 4; ++fn) {
      const _Float16* p = Wt + (size_t)(n0 + 16 * fn + l16) * RANK_ + kc + 8 * half;
      v16h b = pack8(*(const v8h*)p, *(const v8h*)(p + 16));
      acc[0][fn] = wmma_f16(a[0], b, acc[0][fn]);
      acc[1][fn] = wmma_f16(a[1], b, acc[1][fn]);
    }
  }

#pragma unroll
  for (int fm = 0; fm < 2; ++fm)
#pragma unroll
    for (int fn = 0; fn < 4; ++fn)
#pragma unroll
      for (int r = 0; r < 8; ++r) {
        int t = m0 + 16 * fm + r + 8 * half;   // token
        int c = n0 + 16 * fn + l16;            // col in [0,4096)
        float v = acc[fm][fn][r];
        int h = c >> 8, cc = c & 255;
        int b = t >> 11, s = t & 2047;
        if (cc < NOPE_)
          Kbuf[((size_t)(b * H_ + h) * S_ + s) * QK_ + cc] = (_Float16)v;
        else
          Vt[((size_t)(b * H_ + h) * VD_ + (cc - NOPE_)) * S_ + s] = (_Float16)v;
      }
}

// ---------------- causal flash attention ----------------
// grid: B*H*(S/128) blocks, 256 threads = 8 waves, each wave owns 16 q rows.
// K/V kv-tiles (32 rows) are async-staged into double-buffered LDS once per
// block and shared by all 8 waves.
__global__ __launch_bounds__(256) void mla_attn_kernel(
    const float* __restrict__ q, const _Float16* __restrict__ Kbuf,
    const _Float16* __restrict__ Vt, float* __restrict__ out) {
  __shared__ __align__(16) _Float16 Ksh[2][32 * QK_];   // 24 KB
  __shared__ __align__(16) _Float16 Vsh[2][VD_ * 32];   // 16 KB
  __shared__ __align__(16) _Float16 pshm[8][16 * 32];   //  8 KB

  const int tid  = threadIdx.x;
  const int lane = tid & 31;
  const int w    = tid >> 5;
  const int half = lane >> 4;
  const int l16  = lane & 15;
  const int qb = blockIdx.x & 15;      // 16 q-blocks of 128 per (b,h)
  const int bh = blockIdx.x >> 4;
  const int b = bh >> 4, h = bh & 15;
  const int q0 = qb * 128 + w * 16;

  const _Float16* Kh = Kbuf + (size_t)(b * H_ + h) * S_ * QK_;
  const _Float16* Vh = Vt   + (size_t)(b * H_ + h) * VD_ * S_;

  // --- Q fragments: fp32 global -> f16 A-frags, 6 chunks of K=32 ---
  v16h qA[6];
  const float* qrow = q + (size_t)(b * S_ + q0 + l16) * (H_ * QK_) + h * QK_;
#pragma unroll
  for (int c = 0; c < 6; ++c) {
    const float* p0 = qrow + c * 32 + 8 * half;
    v16h a;
#pragma unroll
    for (int j = 0; j < 8; ++j) {
      a[j]     = (_Float16)p0[j];        // k = c*32 + 8*half + j
      a[j + 8] = (_Float16)p0[16 + j];   // k = c*32 + 16 + 8*half + j
    }
    qA[c] = a;
  }

  v8f acc[8] = {};
  float mrow[8], lrow[8];
#pragma unroll
  for (int r = 0; r < 8; ++r) { mrow[r] = -1e30f; lrow[r] = 0.0f; }

  const int ntiles = 4 * qb + 4;       // kv tiles of 32 covering rows <= qb*128+127

  // cooperative async stage of one 32-row kv tile (K: 768 xfers, V: 512 xfers)
  auto stage = [&](int j0, int buf) {
#pragma unroll
    for (int k = 0; k < 3; ++k) {
      int i = tid + 256 * k;           // 0..767
      int row = i / 24, seg = i % 24;  // 24 x 16B segs per 192-f16 K row
      async_ld16(lds_off_of(&Ksh[buf][row * QK_ + seg * 8]),
                 Kh + (size_t)(j0 + row) * QK_ + seg * 8);
    }
#pragma unroll
    for (int k = 0; k < 2; ++k) {
      int i = tid + 256 * k;           // 0..511
      int vd = i >> 2, seg = i & 3;    // 4 x 16B segs per 32-f16 V row
      async_ld16(lds_off_of(&Vsh[buf][vd * 32 + seg * 8]),
                 Vh + (size_t)vd * S_ + j0 + seg * 8);
    }
  };

  stage(0, 0);
  for (int t = 0; t < ntiles; ++t) {
    const int j0 = t * 32;
    wait_async0();                     // my portion of tile t is in LDS
    __syncthreads();                   // everyone's portion; prev reads done
    if (t + 1 < ntiles) stage(j0 + 32, (t + 1) & 1);

    if (j0 <= q0 + 15) {               // wave-uniform causal skip
      const _Float16* ksh = Ksh[t & 1];
      const _Float16* vsh = Vsh[t & 1];

      // S = Q @ K_tile^T (two 16x16 col sub-tiles), B-frags from LDS
      v8f sf[2];
#pragma unroll
      for (int kj = 0; kj < 2; ++kj) {
        v8f s = {};
        const _Float16* kp = ksh + (16 * kj + l16) * QK_ + 8 * half;
#pragma unroll
        for (int c = 0; c < 6; ++c) {
          v16h bfr = pack8(*(const v8h*)(kp + c * 32), *(const v8h*)(kp + c * 32 + 16));
          s = wmma_f16(qA[c], bfr, s);
        }
        sf[kj] = s;
      }

      // scale + causal mask (C layout: row = r+8*half, col = l16)
#pragma unroll
      for (int kj = 0; kj < 2; ++kj)
#pragma unroll
        for (int r = 0; r < 8; ++r) {
          float v = sf[kj][r] * SCALE_;
          int qg = q0 + r + 8 * half;
          int kg = j0 + 16 * kj + l16;
          sf[kj][r] = (kg > qg) ? -1e30f : v;
        }

      // online softmax per row (16-lane row groups)
#pragma unroll
      for (int r = 0; r < 8; ++r) {
        float mx = fmaxf(sf[0][r], sf[1][r]);
#pragma unroll
        for (int d = 1; d < 16; d <<= 1) mx = fmaxf(mx, __shfl_xor(mx, d, 32));
        float mnew = fmaxf(mrow[r], mx);
        float corr = __expf(mrow[r] - mnew);
        float p0 = __expf(sf[0][r] - mnew);
        float p1 = __expf(sf[1][r] - mnew);
        float rs = p0 + p1;
#pragma unroll
        for (int d = 1; d < 16; d <<= 1) rs += __shfl_xor(rs, d, 32);
        lrow[r] = lrow[r] * corr + rs;
        mrow[r] = mnew;
#pragma unroll
        for (int f = 0; f < 8; ++f) acc[f][r] *= corr;
        int ro = (r + 8 * half) * 32;
        pshm[w][ro + l16]      = (_Float16)p0;
        pshm[w][ro + 16 + l16] = (_Float16)p1;
      }

      // P (16x32) C-layout -> A-layout via per-wave LDS (same-wave DS in order)
      v16h pA;
      {
        int base = l16 * 32 + 8 * half;
#pragma unroll
        for (int j = 0; j < 8; ++j) {
          pA[j]     = pshm[w][base + j];        // k = 8*half + j
          pA[j + 8] = pshm[w][base + 16 + j];   // k = 16 + 8*half + j
        }
      }

      // O += P @ V_tile  (V tile in LDS, rows = vd, 32 kv cols)
#pragma unroll
      for (int f = 0; f < 8; ++f) {
        const _Float16* vp = vsh + (16 * f + l16) * 32 + 8 * half;
        v16h vB = pack8(*(const v8h*)vp, *(const v8h*)(vp + 16));
        acc[f] = wmma_f16(pA, vB, acc[f]);
      }
    }
  }

  // normalize and write fp32 output [B*S][H*VD]
#pragma unroll
  for (int r = 0; r < 8; ++r) {
    float inv = 1.0f / lrow[r];
    float* op = out + (size_t)(b * S_ + q0 + r + 8 * half) * (H_ * VD_) + h * VD_ + l16;
#pragma unroll
    for (int f = 0; f < 8; ++f) op[16 * f] = acc[f][r] * inv;
  }
}

// ---------------- launch ----------------
extern "C" void kernel_launch(void* const* d_in, const int* in_sizes, int n_in,
                              void* d_out, int out_size, void* d_ws, size_t ws_size,
                              hipStream_t stream) {
  const float* q      = (const float*)d_in[0];   // (B*S, H*QK)
  const float* latent = (const float*)d_in[1];   // (B*S, RANK+ROPE)
  const float* W      = (const float*)d_in[2];   // (RANK, H*(NOPE+VD))
  float* out = (float*)d_out;

  char* ws = (char*)d_ws;
  const size_t AF_BYTES = (size_t)M_ * RANK_ * 2;          // 4 MiB
  const size_t WT_BYTES = (size_t)N_ * RANK_ * 2;          // 4 MiB
  const size_t K_BYTES  = (size_t)B_ * H_ * S_ * QK_ * 2;  // 24 MiB
  _Float16* Af   = (_Float16*)(ws);
  _Float16* Wt   = (_Float16*)(ws + AF_BYTES);
  _Float16* Kbuf = (_Float16*)(ws + AF_BYTES + WT_BYTES);
  _Float16* Vt   = (_Float16*)(ws + AF_BYTES + WT_BYTES + K_BYTES);

  conv_a_kernel<<<(M_ * RANK_) / 256, 256, 0, stream>>>(latent, Af);
  conv_w_kernel<<<(RANK_ * N_) / 256, 256, 0, stream>>>(W, Wt);
  conv_pe_kernel<<<(M_ * H_ * ROPE_) / 256, 256, 0, stream>>>(latent, Kbuf);
  kv_gemm_kernel<<<dim3(N_ / 128, M_ / 128), 256, 0, stream>>>(Af, Wt, Kbuf, Vt);
  mla_attn_kernel<<<B_ * H_ * (S_ / 128), 256, 0, stream>>>(q, Kbuf, Vt, out);
}
